// SimpleModuleNet_3942779978261
// MI455X (gfx1250) — matrix operand order repaired and checked by reading, not compiled, not run on hardware
//
#include <hip/hip_runtime.h>
#include <hip/hip_fp16.h>

// ---------------------------------------------------------------------------
// SimpleModuleNet on MI455X (gfx1250): all big GEMMs via v_wmma_f32_16x16x32_f16
// Conv A-tiles staged in LDS (async LDS-DMA when available), B from L2.
// Dims: B=128, E=16, C=128, H=W=64, PROJ=512, FC=1024, NANS=2
// ---------------------------------------------------------------------------

typedef _Float16 h16;
typedef __attribute__((ext_vector_type(16))) _Float16 v16h;
typedef __attribute__((ext_vector_type(8)))  _Float16 v8h;
typedef __attribute__((ext_vector_type(8)))  float    v8f;
typedef __attribute__((ext_vector_type(4)))  int      v4i;

union V16 { v16h v; v8h h[2]; };

#define NB   128     // batch
#define NC   128     // channels
#define NE   16      // experts
#define NPJ  512     // proj channels
#define NFC  1024    // fc1 width
#define KCONV 1152   // 9*128 reduction length of 3x3 convs

#if defined(__HIP_DEVICE_COMPILE__) && \
    __has_builtin(__builtin_amdgcn_global_load_async_to_lds_b128) && \
    __has_builtin(__builtin_amdgcn_s_wait_asynccnt)
#define HAVE_ASYNC_LDS 1
#else
#define HAVE_ASYNC_LDS 0
#endif

#if HAVE_ASYNC_LDS
typedef __attribute__((address_space(1))) v4i gas_v4i;   // global
typedef __attribute__((address_space(3))) v4i las_v4i;   // LDS
#endif

__device__ __forceinline__ void stage16(const h16* g, h16* l)
{
#if HAVE_ASYNC_LDS
    __builtin_amdgcn_global_load_async_to_lds_b128((gas_v4i*)g, (las_v4i*)l, 0, 0);
#else
    *(v8h*)l = *(const v8h*)g;
#endif
}

__device__ __forceinline__ void stage_fence()
{
#if HAVE_ASYNC_LDS
    __builtin_amdgcn_s_wait_asynccnt(0);   // my wave's async DMAs landed in LDS
#endif
    __syncthreads();                        // everyone's did
}

// ---------------------------------------------------------------------------
// Weight conversion: [G][cin][3][3] f32 -> [G][tap][cin] f16   (B-matrix rows)
// ---------------------------------------------------------------------------
__global__ __launch_bounds__(256)
void cvt_reorder_kernel(const float* __restrict__ src, h16* __restrict__ dst, int total)
{
    int t = blockIdx.x * 256 + threadIdx.x;
    if (t >= total) return;
    int tap = t % 9;
    int cin = (t / 9) % NC;
    int g   = t / (9 * NC);
    dst[(size_t)g * KCONV + tap * NC + cin] = (h16)src[t];
}

// plain f32 -> f16
__global__ __launch_bounds__(256)
void cvt_plain_kernel(const float* __restrict__ src, h16* __restrict__ dst, int total)
{
    int t = blockIdx.x * 256 + threadIdx.x;
    if (t < total) dst[t] = (h16)src[t];
}

// ---------------------------------------------------------------------------
// Stem1: direct conv 3->128 (K=27) on 64x64 + ReLU + 2x2 maxpool
// out: padded NHWC f16 [B][34][34][128], interior at (1,1)
// ---------------------------------------------------------------------------
__global__ __launch_bounds__(256)
void stem1_pool_kernel(const float* __restrict__ img,     // [B][3][64][64]
                       const float* __restrict__ w1,      // [128][3][3][3]
                       const float* __restrict__ b1,      // [128]
                       h16* __restrict__ pad32)           // [B][34][34][128]
{
    int t = blockIdx.x * 256 + threadIdx.x;               // 128*32*32*128 threads
    int c = t & 127;
    int x = (t >> 7) & 31;
    int y = (t >> 12) & 31;
    int b = t >> 17;

    float w[27];
#pragma unroll
    for (int i = 0; i < 27; ++i) w[i] = w1[c * 27 + i];
    float bias = b1[c];

    float mx = 0.f;                                        // post-ReLU values are >= 0
#pragma unroll
    for (int sy = 0; sy < 2; ++sy)
#pragma unroll
    for (int sx = 0; sx < 2; ++sx) {
        int oy = 2 * y + sy, ox = 2 * x + sx;
        float acc = bias;
#pragma unroll
        for (int cin = 0; cin < 3; ++cin) {
            const float* ip = img + ((size_t)(b * 3 + cin) * 64) * 64;
#pragma unroll
            for (int ky = 0; ky < 3; ++ky) {
                int iy = oy + ky - 1;
                if (iy < 0 || iy >= 64) continue;
#pragma unroll
                for (int kx = 0; kx < 3; ++kx) {
                    int ix = ox + kx - 1;
                    if (ix < 0 || ix >= 64) continue;
                    acc += ip[iy * 64 + ix] * w[cin * 9 + ky * 3 + kx];
                }
            }
        }
        acc = acc > 0.f ? acc : 0.f;
        mx = acc > mx ? acc : mx;
    }
    pad32[((size_t)b * 1156 + (y + 1) * 34 + (x + 1)) * NC + c] = (h16)mx;
}

// ---------------------------------------------------------------------------
// Stem2: 3x3 conv 128->128 at 32x32 via WMMA; ReLU; out NHWC [B][32][32][128]
// Block = 8 waves sharing one (b,y,xt) A-tile in LDS; wave w -> cout tile w*16.
// ---------------------------------------------------------------------------
__global__ __launch_bounds__(256)
void stem2_conv_kernel(const h16* __restrict__ src,       // [B][34][34][128] padded
                       h16* __restrict__ out,             // [B][32][32][128]
                       const h16* __restrict__ w16,       // [128][1152]
                       const float* __restrict__ bias)    // [128]
{
    __shared__ h16 smem[3 * 18 * NC];                      // 13.5 KB

    int tid  = threadIdx.x;
    int lane = tid & 31;
    int wave = tid >> 5;
    int blk  = blockIdx.x;                                 // 128*32*2 blocks
    int xt = blk & 1;
    int y  = (blk >> 1) & 31;
    int b  = blk >> 6;

    // stage rows y..y+2, padded cols xt*16 .. xt*16+17  (3 x 18 x 128 halves)
    const h16* gbase = src + ((size_t)b * 1156 + (size_t)y * 34 + xt * 16) * NC;
    for (int i = tid; i < 864; i += 256) {                 // 864 x 16B
        int r = i / 288, rem = i - r * 288;                // 288 vec8 per row
        stage16(gbase + ((size_t)r * 34 * NC) + rem * 8, smem + (r * 18 * NC) + rem * 8);
    }
    stage_fence();

    int n0 = wave * 16;
    int half = lane >> 4;
    int l = lane & 15;

    v8f acc = {0.f, 0.f, 0.f, 0.f, 0.f, 0.f, 0.f, 0.f};
#pragma unroll
    for (int tap = 0; tap < 9; ++tap) {
        int dy = tap / 3, dx = tap % 3;
        const h16* arow = smem + ((dy * 18) + (l + dx)) * NC;
        const h16* wrow = w16 + (size_t)(n0 + l) * KCONV + tap * NC;
#pragma unroll
        for (int c0 = 0; c0 < NC; c0 += 32) {
            V16 a, bm;
            a.h[0]  = *(const v8h*)(arow + c0 + half * 8);      // ds_load_b128
            a.h[1]  = *(const v8h*)(arow + c0 + half * 8 + 16);
            const h16* wp = wrow + c0 + half * 8;
            bm.h[0] = *(const v8h*)(wp);
            bm.h[1] = *(const v8h*)(wp + 16);
            acc = __builtin_amdgcn_wmma_f32_16x16x32_f16(false, a.v, false, bm.v,
                                                         (short)0, acc, false, false);
        }
    }
    float bv = bias[n0 + l];
#pragma unroll
    for (int i = 0; i < 8; ++i) {
        int mm = i + half * 8;                             // spatial x within tile
        float v = acc[i] + bv;
        v = v > 0.f ? v : 0.f;
        out[((size_t)(b * 32 + y) * 32 + xt * 16 + mm) * NC + n0 + l] = (h16)v;
    }
}

// 2x2 maxpool 32x32 -> padded [B][18][18][128]
__global__ __launch_bounds__(256)
void pool32_kernel(const h16* __restrict__ src, h16* __restrict__ x18)
{
    int t = blockIdx.x * 256 + threadIdx.x;                // 128*16*16*128
    int c = t & 127;
    int x = (t >> 7) & 15;
    int y = (t >> 11) & 15;
    int b = t >> 15;
    float m = -3.4e38f;
#pragma unroll
    for (int sy = 0; sy < 2; ++sy)
#pragma unroll
    for (int sx = 0; sx < 2; ++sx) {
        float v = (float)src[((size_t)(b * 32 + 2 * y + sy) * 32 + 2 * x + sx) * NC + c];
        m = v > m ? v : m;
    }
    x18[((size_t)b * 324 + (y + 1) * 18 + (x + 1)) * NC + c] = (h16)m;
}

// ---------------------------------------------------------------------------
// Expert block conv (3x3, 128->128 at 16x16), per-sample expert weights.
// Block = 8 waves sharing one (b,y) A-row in LDS; wave w -> cout tile w*16.
// phase 0: dst = relu(conv(src)+b)                  (src=x18 -> dst=t18)
// phase 1: dst = relu(xres + conv(src)+b)           (src=t18 -> dst=xres=x18)
// ---------------------------------------------------------------------------
__global__ __launch_bounds__(256)
void expert_conv_kernel(const h16* __restrict__ src,      // padded [B][18][18][128]
                        h16* __restrict__ dst,            // padded [B][18][18][128]
                        const h16* __restrict__ xres,     // residual (phase 1)
                        const h16* __restrict__ w16,      // [E][128][1152]
                        const float* __restrict__ bias,   // [E][128]
                        const int* __restrict__ question, // [B][8]
                        int col, int phase)
{
    __shared__ h16 smem[3 * 18 * NC];                      // 13.5 KB

    int tid  = threadIdx.x;
    int lane = tid & 31;
    int wave = tid >> 5;
    int blk  = blockIdx.x;                                 // 128*16 blocks
    int y = blk & 15;
    int b = blk >> 4;

    // stage padded rows y..y+2 (full 18-wide rows are contiguous)
    const h16* gbase = src + ((size_t)b * 324 + (size_t)y * 18) * NC;
    for (int i = tid; i < 864; i += 256)                   // 864 x 16B contiguous
        stage16(gbase + i * 8, smem + i * 8);
    stage_fence();

    int n0 = wave * 16;
    int half = lane >> 4;
    int l = lane & 15;

    int e = question[b * 8 + col];
    const h16*  wb = w16  + (size_t)e * NC * KCONV;
    const float* bb = bias + e * NC;

    v8f acc = {0.f, 0.f, 0.f, 0.f, 0.f, 0.f, 0.f, 0.f};
#pragma unroll
    for (int tap = 0; tap < 9; ++tap) {
        int dy = tap / 3, dx = tap % 3;
        const h16* arow = smem + ((dy * 18) + (l + dx)) * NC;
        const h16* wrow = wb + (size_t)(n0 + l) * KCONV + tap * NC;
#pragma unroll
        for (int c0 = 0; c0 < NC; c0 += 32) {
            V16 a, bm;
            a.h[0]  = *(const v8h*)(arow + c0 + half * 8);      // ds_load_b128
            a.h[1]  = *(const v8h*)(arow + c0 + half * 8 + 16);
            const h16* wp = wrow + c0 + half * 8;
            bm.h[0] = *(const v8h*)(wp);
            bm.h[1] = *(const v8h*)(wp + 16);
            acc = __builtin_amdgcn_wmma_f32_16x16x32_f16(false, a.v, false, bm.v,
                                                         (short)0, acc, false, false);
        }
    }
    float bv = bb[n0 + l];
    h16* dbase = dst + (size_t)b * 324 * NC;
    const h16* rbase = xres + (size_t)b * 324 * NC;
#pragma unroll
    for (int i = 0; i < 8; ++i) {
        int mm = i + half * 8;
        size_t off = ((size_t)(y + 1) * 18 + (mm + 1)) * NC + n0 + l;
        float v = acc[i] + bv;
        if (phase == 1) v += (float)rbase[off];
        v = v > 0.f ? v : 0.f;
        dbase[off] = (h16)v;
    }
}

// ---------------------------------------------------------------------------
// Proj: 1x1 conv 128->512 at 16x16 via WMMA, ReLU -> [B][16][16][512]
// ---------------------------------------------------------------------------
__global__ __launch_bounds__(128)
void proj_conv_kernel(const h16* __restrict__ x18,        // padded [B][18][18][128]
                      h16* __restrict__ out,              // [B][16][16][512]
                      const h16* __restrict__ w16,        // [512][128]
                      const float* __restrict__ bias)     // [512]
{
    int lane = threadIdx.x & 31;
    int t = (blockIdx.x * blockDim.x + threadIdx.x) >> 5;
    int n0 = (t & 31) * 16;
    int y  = (t >> 5) & 15;
    int b  = t >> 9;
    int half = lane >> 4;
    int l = lane & 15;

    const h16* arow = x18 + ((size_t)b * 324 + (y + 1) * 18 + (l + 1)) * NC;
    const h16* wrow = w16 + (size_t)(n0 + l) * NC;
    v8f acc = {0.f, 0.f, 0.f, 0.f, 0.f, 0.f, 0.f, 0.f};

#pragma unroll
    for (int c0 = 0; c0 < NC; c0 += 32) {
        V16 a, bm;
        a.h[0]  = *(const v8h*)(arow + c0 + half * 8);
        a.h[1]  = *(const v8h*)(arow + c0 + half * 8 + 16);
        const h16* wp = wrow + c0 + half * 8;
        bm.h[0] = *(const v8h*)(wp);
        bm.h[1] = *(const v8h*)(wp + 16);
        acc = __builtin_amdgcn_wmma_f32_16x16x32_f16(false, a.v, false, bm.v,
                                                     (short)0, acc, false, false);
    }
    float bv = bias[n0 + l];
#pragma unroll
    for (int i = 0; i < 8; ++i) {
        int mm = i + half * 8;
        float v = acc[i] + bv;
        v = v > 0.f ? v : 0.f;
        out[((size_t)(b * 16 + y) * 16 + mm) * NPJ + n0 + l] = (h16)v;
    }
}

// maxpool 16x16->8x8 and scatter to NCHW-flatten order: feat[b][c*64 + y*8 + x]
__global__ __launch_bounds__(256)
void proj_pool_kernel(const h16* __restrict__ src, h16* __restrict__ feat)
{
    int t = blockIdx.x * 256 + threadIdx.x;                // 128*512*8*8
    int c = t & 511;
    int x = (t >> 9) & 7;
    int y = (t >> 12) & 7;
    int b = t >> 15;
    float m = -3.4e38f;
#pragma unroll
    for (int sy = 0; sy < 2; ++sy)
#pragma unroll
    for (int sx = 0; sx < 2; ++sx) {
        float v = (float)src[((size_t)(b * 16 + 2 * y + sy) * 16 + 2 * x + sx) * NPJ + c];
        m = v > m ? v : m;
    }
    feat[(size_t)b * 32768 + c * 64 + y * 8 + x] = (h16)m;
}

// ---------------------------------------------------------------------------
// FC1: [128 x 32768] @ [32768 x 1024] via WMMA, ReLU -> f16 [128][1024]
// ---------------------------------------------------------------------------
__global__ __launch_bounds__(128)
void fc1_kernel(const h16* __restrict__ feat,             // [128][32768]
                const h16* __restrict__ w16,              // [1024][32768]
                const float* __restrict__ bias,           // [1024]
                h16* __restrict__ out)                    // [128][1024]
{
    int lane = threadIdx.x & 31;
    int t = (blockIdx.x * blockDim.x + threadIdx.x) >> 5;
    int nt = t & 63;
    int mt = t >> 6;
    int m0 = mt * 16, n0 = nt * 16;
    int half = lane >> 4;
    int l = lane & 15;

    const h16* arow = feat + (size_t)(m0 + l) * 32768;
    const h16* wrow = w16  + (size_t)(n0 + l) * 32768;
    v8f acc = {0.f, 0.f, 0.f, 0.f, 0.f, 0.f, 0.f, 0.f};

    for (int k0 = 0; k0 < 32768; k0 += 32) {
        __builtin_prefetch(wrow + k0 + 1024, 0, 1);        // global_prefetch_b8
        V16 a, bm;
        a.h[0]  = *(const v8h*)(arow + k0 + half * 8);
        a.h[1]  = *(const v8h*)(arow + k0 + half * 8 + 16);
        const h16* wp = wrow + k0 + half * 8;
        bm.h[0] = *(const v8h*)(wp);
        bm.h[1] = *(const v8h*)(wp + 16);
        acc = __builtin_amdgcn_wmma_f32_16x16x32_f16(false, a.v, false, bm.v,
                                                     (short)0, acc, false, false);
    }
    float bv = bias[n0 + l];
#pragma unroll
    for (int i = 0; i < 8; ++i) {
        int mm = i + half * 8;
        float v = acc[i] + bv;
        v = v > 0.f ? v : 0.f;
        out[(size_t)(m0 + mm) * NFC + n0 + l] = (h16)v;
    }
}

// FC2: [128 x 1024] @ [1024 x 2] + bias -> f32 output
__global__ __launch_bounds__(256)
void fc2_kernel(const h16* __restrict__ f,                // [128][1024]
                const float* __restrict__ w,              // [2][1024]
                const float* __restrict__ bias,           // [2]
                float* __restrict__ out)                  // [128][2]
{
    int t = blockIdx.x * 256 + threadIdx.x;                // 256 threads
    if (t >= NB * 2) return;
    int o = t & 1;
    int b = t >> 1;
    float s = bias[o];
    const h16* fr = f + (size_t)b * NFC;
    const float* wr = w + (size_t)o * NFC;
    for (int k = 0; k < NFC; ++k) s += (float)fr[k] * wr[k];
    out[b * 2 + o] = s;
}

// ---------------------------------------------------------------------------
extern "C" void kernel_launch(void* const* d_in, const int* in_sizes, int n_in,
                              void* d_out, int out_size, void* d_ws, size_t ws_size,
                              hipStream_t stream) {
    const float* image    = (const float*)d_in[0];
    const int*   question = (const int*)  d_in[1];
    const float* stem_w1  = (const float*)d_in[2];
    const float* stem_b1  = (const float*)d_in[3];
    const float* stem_w2  = (const float*)d_in[4];
    const float* stem_b2  = (const float*)d_in[5];
    const float* exp_w1   = (const float*)d_in[6];
    const float* exp_b1   = (const float*)d_in[7];
    const float* exp_w2   = (const float*)d_in[8];
    const float* exp_b2   = (const float*)d_in[9];
    const float* proj_w   = (const float*)d_in[10];
    const float* proj_b   = (const float*)d_in[11];
    const float* fc1_w    = (const float*)d_in[12];
    const float* fc1_b    = (const float*)d_in[13];
    const float* fc2_w    = (const float*)d_in[14];
    const float* fc2_b    = (const float*)d_in[15];
    float* out = (float*)d_out;
    char*  ws  = (char*)d_ws;

    // ---- workspace arena (regions reused across pipeline stages) ----
    const size_t SZ_PAD32  = (size_t)NB * 34 * 34 * NC * 2;   // 37.9 MB
    const size_t SZ_FC1W16 = (size_t)NFC * 32768 * 2;         // 64  MB
    const size_t SZ_C32OUT = (size_t)NB * 32 * 32 * NC * 2;   // 33.5 MB (== proj out size)
    const size_t SZ_P18    = (size_t)NB * 18 * 18 * NC * 2;   // 10.6 MB
    const size_t SZ_FEAT   = (size_t)NB * 32768 * 2;          // 8.4 MB
    const size_t SZ_FC1OUT = (size_t)NB * NFC * 2;
    const size_t SZ_W2     = (size_t)NC * KCONV * 2;
    const size_t SZ_EXPW   = (size_t)NE * NC * KCONV * 2;
    const size_t SZ_PROJW  = (size_t)NPJ * NC * 2;

    size_t off = 0;
    const size_t R0 = off; off += SZ_FC1W16;     // pad32 early, fc1 f16 weights late
    const size_t R1 = off; off += SZ_C32OUT;     // stem2 conv out early, proj out late
    const size_t R2 = off; off += SZ_P18;        // x18
    const size_t R3 = off; off += SZ_P18;        // t18
    const size_t R4 = off; off += SZ_FEAT;       // feat
    const size_t R5 = off; off += SZ_FC1OUT;     // fc1 out
    const size_t R6 = off; off += SZ_W2;         // stem2 w f16
    const size_t R7 = off; off += SZ_EXPW;       // exp_w1 f16
    const size_t R8 = off; off += SZ_EXPW;       // exp_w2 f16
    const size_t R9 = off; off += SZ_PROJW;      // proj w f16

    h16* pad32   = (h16*)(ws + R0);
    h16* fc1w16  = (h16*)(ws + R0);
    h16* c32out  = (h16*)(ws + R1);
    h16* projout = (h16*)(ws + R1);
    h16* x18     = (h16*)(ws + R2);
    h16* t18     = (h16*)(ws + R3);
    h16* feat    = (h16*)(ws + R4);
    h16* fc1out  = (h16*)(ws + R5);
    h16* w2_16   = (h16*)(ws + R6);
    h16* ew1_16  = (h16*)(ws + R7);
    h16* ew2_16  = (h16*)(ws + R8);
    h16* pw16    = (h16*)(ws + R9);

    // zero the padded borders (interiors get fully overwritten)
    (void)hipMemsetAsync(pad32, 0, SZ_PAD32, stream);
    (void)hipMemsetAsync(x18,   0, SZ_P18,   stream);
    (void)hipMemsetAsync(t18,   0, SZ_P18,   stream);

    // ---- weight conversions (small ones; fc1 conversion deferred) ----
    {
        int n = NC * NC * 9;
        cvt_reorder_kernel<<<(n + 255) / 256, 256, 0, stream>>>(stem_w2, w2_16, n);
        n = NE * NC * NC * 9;
        cvt_reorder_kernel<<<(n + 255) / 256, 256, 0, stream>>>(exp_w1, ew1_16, n);
        cvt_reorder_kernel<<<(n + 255) / 256, 256, 0, stream>>>(exp_w2, ew2_16, n);
        n = NPJ * NC;
        cvt_plain_kernel<<<(n + 255) / 256, 256, 0, stream>>>(proj_w, pw16, n);
    }

    // ---- stem ----
    stem1_pool_kernel<<<(NB * 32 * 32 * NC) / 256, 256, 0, stream>>>(image, stem_w1, stem_b1, pad32);
    stem2_conv_kernel<<<NB * 32 * 2, 256, 0, stream>>>(pad32, c32out, w2_16, stem_b2);
    pool32_kernel<<<(NB * 16 * 16 * NC) / 256, 256, 0, stream>>>(c32out, x18);

    // pad32 is now dead -> convert fc1 weights into the same region
    {
        int n = NFC * 32768;
        cvt_plain_kernel<<<(n + 255) / 256, 256, 0, stream>>>(fc1_w, fc1w16, n);
    }

    // ---- 5 expert residual blocks ----
    const int cols[5] = {3, 4, 6, 7, 5};
    for (int i = 0; i < 5; ++i) {
        expert_conv_kernel<<<NB * 16, 256, 0, stream>>>(x18, t18, x18, ew1_16, exp_b1,
                                                        question, cols[i], 0);
        expert_conv_kernel<<<NB * 16, 256, 0, stream>>>(t18, x18, x18, ew2_16, exp_b2,
                                                        question, cols[i], 1);
    }

    // ---- proj + pool + flatten ----
    proj_conv_kernel<<<(NB * 16 * 32) / 4, 128, 0, stream>>>(x18, projout, pw16, proj_b);
    proj_pool_kernel<<<(NB * NPJ * 8 * 8) / 256, 256, 0, stream>>>(projout, feat);

    // ---- fully connected ----
    fc1_kernel<<<(8 * 64) / 4, 128, 0, stream>>>(feat, fc1w16, fc1_b, fc1out);
    fc2_kernel<<<1, 256, 0, stream>>>(fc1out, fc2_w, fc2_b, out);

    (void)in_sizes; (void)n_in; (void)out_size; (void)ws_size;
}
